// GATNetSelectiveResidualsUpdated_31628139168035
// MI455X (gfx1250) — matrix operand order repaired and checked by reading, not compiled, not run on hardware
//
#include <hip/hip_runtime.h>
#include <stdint.h>

// ---------- problem constants ----------
#define KN   12288      // nodes
#define KE   294912     // edges (before self loops)
#define KET  (KE + KN)  // edges + self loops
#define KD   512        // input dim / H*C

typedef __attribute__((ext_vector_type(16))) __bf16 v16bf;
typedef __attribute__((ext_vector_type(8)))  float  v8f;

// ---------- helpers ----------
__device__ __forceinline__ unsigned short f2bf(float f) {
  union { float f; uint32_t u; } c; c.f = f;
  uint32_t u = c.u;
  uint32_t r = (u + 0x7FFFu + ((u >> 16) & 1u)) >> 16;   // round-to-nearest-even
  return (unsigned short)r;
}

__device__ __forceinline__ void atomicMaxFloat(float* addr, float val) {
  if (val >= 0.0f) atomicMax((int*)addr, __float_as_int(val));
  else             atomicMin((unsigned int*)addr, (unsigned int)__float_as_int(val));
}

// A-fragment (16x32 bf16): lane = M row, kg = lane>>4 selects K half-group
__device__ __forceinline__ void load_frag_a(const unsigned short* __restrict__ base,
                                            int k0, int kg, v16bf& a) {
  #pragma unroll
  for (int v = 0; v < 8; ++v) {
    int koff = ((v < 4) ? (v * 2) : (16 + (v - 4) * 2)) + kg * 8;
    union { uint32_t u; __bf16 h[2]; } c;
    c.u = *(const uint32_t*)(base + k0 + koff);
    a[2 * v] = c.h[0]; a[2 * v + 1] = c.h[1];
  }
}

// B-fragment (32x16 bf16): lane = N col, kg = lane>>4 selects K half-group (same K order as A)
__device__ __forceinline__ void load_frag_b(const unsigned short* __restrict__ base,
                                            int k0, int kg, v16bf& b) {
  #pragma unroll
  for (int v = 0; v < 8; ++v) {
    int koff = ((v < 4) ? (v * 2) : (16 + (v - 4) * 2)) + kg * 8;
    union { uint32_t u; __bf16 h[2]; } c;
    c.u = *(const uint32_t*)(base + k0 + koff);
    b[2 * v] = c.h[0]; b[2 * v + 1] = c.h[1];
  }
}

// ---------- generic fills / converts ----------
__global__ void fill_f32(float* __restrict__ p, float v, size_t n) {
  size_t i = (size_t)blockIdx.x * blockDim.x + threadIdx.x;
  if (i < n) p[i] = v;
}

__global__ void cvt_bf16(const float* __restrict__ X, unsigned short* __restrict__ Xb, size_t n) {
  size_t i = (size_t)blockIdx.x * blockDim.x + threadIdx.x;
  if (i < n) Xb[i] = f2bf(X[i]);
}

// W [K x N] f32 row-major  ->  WT [N x K] bf16 (B-matrix K contiguous per lane)
__global__ void transpose_bf16(const float* __restrict__ W, unsigned short* __restrict__ WT,
                               int K, int Nn) {
  int idx = blockIdx.x * blockDim.x + threadIdx.x;
  if (idx >= K * Nn) return;
  int k = idx / Nn, n = idx - k * Nn;
  WT[(size_t)n * K + k] = f2bf(W[(size_t)k * Nn + n]);
}

// ---------- WMMA GEMM: Out[M,N] = A(bf16)[M,K] * BT(bf16)[N,K]^T + bias ----------
// wave computes a 32x64 tile: 2 A-fragments x 4 B-fragments -> 8 wmma / K-step
// (3:1 fragment-load : wmma ratio vs 8:1 for a single-tile wave)
__global__ __launch_bounds__(256)
void wmma_gemm_bias(const unsigned short* __restrict__ A,
                    const unsigned short* __restrict__ BT,
                    const float* __restrict__ bias,
                    float* __restrict__ Out,
                    int M, int Nn, int K) {
  const int lane = threadIdx.x & 31;
  const int wave = threadIdx.x >> 5;
  const int n0 = blockIdx.x * 64;
  const int m0 = (blockIdx.y * 8 + wave) * 32;
  if (m0 >= M) return;                     // wave-uniform: EXEC stays all-ones for WMMA

  const int row = lane & 15;
  const int kg  = lane >> 4;
  const unsigned short* Arow0 = A + (size_t)(m0 + row) * K;
  const unsigned short* Arow1 = A + (size_t)(m0 + 16 + row) * K;
  const unsigned short* Bcol[4];
  #pragma unroll
  for (int j = 0; j < 4; ++j) Bcol[j] = BT + (size_t)(n0 + j * 16 + row) * K;

  v8f acc[2][4] = {};
  for (int k0 = 0; k0 < K; k0 += 32) {
    __builtin_prefetch(Arow0 + k0 + 128, 0, 3);   // global_prefetch_b8 (near)
    __builtin_prefetch(Arow1 + k0 + 128, 0, 3);
    v16bf a0, a1, b[4];
    load_frag_a(Arow0, k0, kg, a0);
    load_frag_a(Arow1, k0, kg, a1);
    #pragma unroll
    for (int j = 0; j < 4; ++j) load_frag_b(Bcol[j], k0, kg, b[j]);
    #pragma unroll
    for (int j = 0; j < 4; ++j) {
      acc[0][j] = __builtin_amdgcn_wmma_f32_16x16x32_bf16(false, a0, false, b[j],
                                                          (short)0, acc[0][j], false, false);
      acc[1][j] = __builtin_amdgcn_wmma_f32_16x16x32_bf16(false, a1, false, b[j],
                                                          (short)0, acc[1][j], false, false);
    }
  }
  // C/D layout: lane -> col = (lane&15); VGPR r -> row = r + 8*(lane>>4)
  #pragma unroll
  for (int tj = 0; tj < 4; ++tj) {
    const int col = n0 + tj * 16 + (lane & 15);
    const float bv = bias ? bias[col] : 0.0f;
    #pragma unroll
    for (int ti = 0; ti < 2; ++ti) {
      #pragma unroll
      for (int r = 0; r < 8; ++r) {
        int mrow = m0 + ti * 16 + r + kg * 8;
        Out[(size_t)mrow * Nn + col] = acc[ti][tj][r] + bv;
      }
    }
  }
}

// ---------- GATv2 edge phase ----------
// pass 1: e[edge,h] = sum_c leakyrelu(xl[src,c]+xr[dst,c])*att[c]; atomicMax per dst
__global__ __launch_bounds__(256)
void edge_scores(const int* __restrict__ ei, const float* __restrict__ xl,
                 const float* __restrict__ xr, const float* __restrict__ att,
                 float* __restrict__ e, float* __restrict__ emax) {
  int edge = blockIdx.x * 8 + (threadIdx.x >> 5);
  if (edge >= KET) return;
  int lane = threadIdx.x & 31;
  int s, d;
  if (edge < KE) { s = ei[edge]; d = ei[KE + edge]; } else { s = d = edge - KE; }
  const float* pl = xl + (size_t)s * KD;
  const float* pr = xr + (size_t)d * KD;
  float acc = 0.0f;
  int c0 = lane * 16;                      // lanes 0-15 -> head 0, lanes 16-31 -> head 1
  #pragma unroll
  for (int t = 0; t < 16; ++t) {
    int c = c0 + t;
    float m = pl[c] + pr[c];
    m = (m > 0.0f) ? m : 0.2f * m;
    acc += m * att[c];
  }
  #pragma unroll
  for (int off = 1; off < 16; off <<= 1) acc += __shfl_xor(acc, off, 32);
  if ((lane & 15) == 0) {
    int h = lane >> 4;
    e[(size_t)edge * 2 + h] = acc;
    atomicMaxFloat(&emax[(size_t)d * 2 + h], acc);
  }
}

// pass 2: z = exp(e - emax[dst]); denom[dst] += z
__global__ void edge_softmax_z(const int* __restrict__ ei, const float* __restrict__ e,
                               const float* __restrict__ emax,
                               float* __restrict__ z, float* __restrict__ denom) {
  int idx = blockIdx.x * blockDim.x + threadIdx.x;
  if (idx >= KET * 2) return;
  int edge = idx >> 1, h = idx & 1;
  int d = (edge < KE) ? ei[KE + edge] : (edge - KE);
  float zz = __expf(e[idx] - emax[(size_t)d * 2 + h]);
  z[idx] = zz;
  atomicAdd(&denom[(size_t)d * 2 + h], zz);
}

// pass 3: gat[dst] += (z/denom[dst]) * xl[src]
__global__ __launch_bounds__(256)
void edge_aggregate(const int* __restrict__ ei, const float* __restrict__ xl,
                    const float* __restrict__ z, const float* __restrict__ denom,
                    float* __restrict__ gat) {
  int edge = blockIdx.x * 8 + (threadIdx.x >> 5);
  if (edge >= KET) return;
  int lane = threadIdx.x & 31;
  int s, d;
  if (edge < KE) { s = ei[edge]; d = ei[KE + edge]; } else { s = d = edge - KE; }
  int h = lane >> 4;
  float alpha = z[(size_t)edge * 2 + h] / denom[(size_t)d * 2 + h];
  const float* pl = xl + (size_t)s * KD;
  float* pg = gat + (size_t)d * KD;
  int c0 = lane * 16;
  #pragma unroll
  for (int t = 0; t < 16; ++t) {
    int c = c0 + t;
    atomicAdd(&pg[c], alpha * pl[c]);
  }
}

// h = relu(gat + bias_c) -> bf16 activation for next GEMM
__global__ void relu_bias_bf16(const float* __restrict__ gat, const float* __restrict__ bias,
                               unsigned short* __restrict__ hb, size_t total) {
  size_t i = (size_t)blockIdx.x * blockDim.x + threadIdx.x;
  if (i >= total) return;
  float v = gat[i] + bias[i & (KD - 1)];
  hb[i] = f2bf(v > 0.0f ? v : 0.0f);
}

// ---------- fused LayerNorm + ReLU (+ residual), one wave per row ----------
__global__ __launch_bounds__(256)
void ln_relu_res(const float* __restrict__ pre, const float* __restrict__ res,
                 const float* __restrict__ g, const float* __restrict__ b,
                 float* __restrict__ outf, unsigned short* __restrict__ outb,
                 int D, int rows) {
  int row = blockIdx.x * 8 + (threadIdx.x >> 5);
  if (row >= rows) return;
  int lane = threadIdx.x & 31;
  const float* p = pre + (size_t)row * D;
  int per = D >> 5;                        // D in {64,128,256} -> 2/4/8 per lane
  float vals[8];
  float sum = 0.0f;
  for (int t = 0; t < per; ++t) { vals[t] = p[lane * per + t]; sum += vals[t]; }
  #pragma unroll
  for (int off = 1; off < 32; off <<= 1) sum += __shfl_xor(sum, off, 32);
  float mu = sum / (float)D;
  float var = 0.0f;
  for (int t = 0; t < per; ++t) { float dd = vals[t] - mu; var += dd * dd; }
  #pragma unroll
  for (int off = 1; off < 32; off <<= 1) var += __shfl_xor(var, off, 32);
  float rstd = rsqrtf(var / (float)D + 1e-5f);
  for (int t = 0; t < per; ++t) {
    int c = lane * per + t;
    float v = (vals[t] - mu) * rstd * g[c] + b[c];
    v = (v > 0.0f) ? v : 0.0f;
    if (res)  v += res[(size_t)row * D + c];
    if (outf) outf[(size_t)row * D + c] = v;
    if (outb) outb[(size_t)row * D + c] = f2bf(v);
  }
}

// ---------- tiny final GEMM 64 -> 3 ----------
__global__ void gemm64x3(const float* __restrict__ H3, const float* __restrict__ W,
                         const float* __restrict__ bb, float* __restrict__ Y, int M) {
  int i = blockIdx.x * blockDim.x + threadIdx.x;
  if (i >= M) return;
  float a0 = bb[0], a1 = bb[1], a2 = bb[2];
  const float* h = H3 + (size_t)i * 64;
  #pragma unroll 4
  for (int k = 0; k < 64; ++k) {
    float hv = h[k];
    a0 += hv * W[k * 3 + 0];
    a1 += hv * W[k * 3 + 1];
    a2 += hv * W[k * 3 + 2];
  }
  Y[(size_t)i * 3 + 0] = a0;
  Y[(size_t)i * 3 + 1] = a1;
  Y[(size_t)i * 3 + 2] = a2;
}

// ---------- cdist: out[i,j] = sqrt(max(|yi|^2+|yj|^2-2 yi.yj, 0)) ----------
__global__ __launch_bounds__(256)
void cdist_kernel(const float* __restrict__ Y, float* __restrict__ out, int Nn) {
  __shared__ float yi[16][4];
  __shared__ float yj[16][4];
  int tx = threadIdx.x, ty = threadIdx.y;
  int t = ty * 16 + tx;
  if (t < 48)      yi[t / 3][t % 3] = Y[((size_t)blockIdx.y * 16 + t / 3) * 3 + t % 3];
  else if (t < 96) { int u = t - 48; yj[u / 3][u % 3] = Y[((size_t)blockIdx.x * 16 + u / 3) * 3 + u % 3]; }
  __syncthreads();
  float a0 = yi[ty][0], a1 = yi[ty][1], a2 = yi[ty][2];
  float b0 = yj[tx][0], b1 = yj[tx][1], b2 = yj[tx][2];
  float si = a0 * a0 + a1 * a1 + a2 * a2;
  float sj = b0 * b0 + b1 * b1 + b2 * b2;
  float dot = a0 * b0 + a1 * b1 + a2 * b2;
  float d2 = si + sj - 2.0f * dot;
  d2 = d2 > 0.0f ? d2 : 0.0f;
  float dst = (d2 > 0.0f) ? sqrtf(d2) : 0.0f;
  int i = blockIdx.y * 16 + ty;
  int j = blockIdx.x * 16 + tx;
  out[(size_t)i * Nn + j] = dst;
}

// ---------- host launcher ----------
extern "C" void kernel_launch(void* const* d_in, const int* in_sizes, int n_in,
                              void* d_out, int out_size, void* d_ws, size_t ws_size,
                              hipStream_t stream) {
  (void)in_sizes; (void)n_in; (void)out_size; (void)ws_size;
  const float* x      = (const float*)d_in[0];
  const int*   ei     = (const int*)  d_in[1];
  const float* Wl     = (const float*)d_in[2];
  const float* bl     = (const float*)d_in[3];
  const float* Wr     = (const float*)d_in[4];
  const float* br     = (const float*)d_in[5];
  const float* att    = (const float*)d_in[6];
  const float* bias_c = (const float*)d_in[7];
  const float* Wd_a   = (const float*)d_in[8];
  const float* bd_a   = (const float*)d_in[9];
  const float* Wal_a  = (const float*)d_in[10];
  const float* bal_a  = (const float*)d_in[11];
  const float* ga     = (const float*)d_in[12];
  const float* bta    = (const float*)d_in[13];
  const float* Wd1    = (const float*)d_in[14];
  const float* bd1    = (const float*)d_in[15];
  const float* Wal1   = (const float*)d_in[16];
  const float* bal1   = (const float*)d_in[17];
  const float* g1     = (const float*)d_in[18];
  const float* bt1    = (const float*)d_in[19];
  const float* Wd2    = (const float*)d_in[20];
  const float* bd2    = (const float*)d_in[21];
  const float* g2     = (const float*)d_in[22];
  const float* bt2    = (const float*)d_in[23];
  const float* Wd3    = (const float*)d_in[24];
  const float* bd3    = (const float*)d_in[25];
  float* out = (float*)d_out;

  // bump allocator over workspace
  char* ws = (char*)d_ws;
  size_t off = 0;
  auto alloc = [&](size_t bytes) -> void* {
    void* p = ws + off;
    off = (off + bytes + 255) & ~(size_t)255;
    return p;
  };

  unsigned short* xb    = (unsigned short*)alloc((size_t)KN * KD * 2);
  unsigned short* WlT   = (unsigned short*)alloc((size_t)512 * 512 * 2);
  unsigned short* WrT   = (unsigned short*)alloc((size_t)512 * 512 * 2);
  unsigned short* WdaT  = (unsigned short*)alloc((size_t)256 * 512 * 2);
  unsigned short* WalaT = (unsigned short*)alloc((size_t)256 * 512 * 2);
  unsigned short* Wd1T  = (unsigned short*)alloc((size_t)128 * 256 * 2);
  unsigned short* Wal1T = (unsigned short*)alloc((size_t)128 * 256 * 2);
  unsigned short* Wd2T  = (unsigned short*)alloc((size_t)64  * 128 * 2);
  float* xl    = (float*)alloc((size_t)KN * KD * 4);
  float* xr    = (float*)alloc((size_t)KN * KD * 4);
  float* e     = (float*)alloc((size_t)KET * 2 * 4);
  float* z     = (float*)alloc((size_t)KET * 2 * 4);
  float* emax  = (float*)alloc((size_t)KN * 2 * 4);
  float* denom = (float*)alloc((size_t)KN * 2 * 4);
  float* gat   = (float*)alloc((size_t)KN * KD * 4);
  unsigned short* hb  = (unsigned short*)alloc((size_t)KN * KD * 2);
  float* t1 = (float*)alloc((size_t)KN * 256 * 4);
  float* r1 = (float*)alloc((size_t)KN * 256 * 4);
  unsigned short* h1b = (unsigned short*)alloc((size_t)KN * 256 * 2);
  float* t2 = (float*)alloc((size_t)KN * 128 * 4);
  float* r2 = (float*)alloc((size_t)KN * 128 * 4);
  unsigned short* h2b = (unsigned short*)alloc((size_t)KN * 128 * 2);
  float* t3  = (float*)alloc((size_t)KN * 64 * 4);
  float* h3f = (float*)alloc((size_t)KN * 64 * 4);
  float* y   = (float*)alloc((size_t)KN * 3 * 4);

  // 1. precision conversion / weight pre-transpose (bf16)
  cvt_bf16<<<(KN * KD + 255) / 256, 256, 0, stream>>>(x, xb, (size_t)KN * KD);
  transpose_bf16<<<(512 * 512 + 255) / 256, 256, 0, stream>>>(Wl,    WlT,   512, 512);
  transpose_bf16<<<(512 * 512 + 255) / 256, 256, 0, stream>>>(Wr,    WrT,   512, 512);
  transpose_bf16<<<(512 * 256 + 255) / 256, 256, 0, stream>>>(Wd_a,  WdaT,  512, 256);
  transpose_bf16<<<(512 * 256 + 255) / 256, 256, 0, stream>>>(Wal_a, WalaT, 512, 256);
  transpose_bf16<<<(256 * 128 + 255) / 256, 256, 0, stream>>>(Wd1,   Wd1T,  256, 128);
  transpose_bf16<<<(256 * 128 + 255) / 256, 256, 0, stream>>>(Wal1,  Wal1T, 256, 128);
  transpose_bf16<<<(128 * 64  + 255) / 256, 256, 0, stream>>>(Wd2,   Wd2T,  128, 64);

  // 2. xl = x@Wl + bl ; xr = x@Wr + br   (WMMA bf16, 32x64 tile per wave)
  dim3 g512(512 / 64, KN / 256);
  wmma_gemm_bias<<<g512, 256, 0, stream>>>(xb, WlT, bl, xl, KN, 512, 512);
  wmma_gemm_bias<<<g512, 256, 0, stream>>>(xb, WrT, br, xr, KN, 512, 512);

  // 3. edge softmax + aggregation (segment buffers resolve in L2)
  fill_f32<<<(KN * 2 + 255) / 256, 256, 0, stream>>>(emax, -INFINITY, (size_t)KN * 2);
  fill_f32<<<(KN * 2 + 255) / 256, 256, 0, stream>>>(denom, 0.0f, (size_t)KN * 2);
  fill_f32<<<(KN * KD + 255) / 256, 256, 0, stream>>>(gat, 0.0f, (size_t)KN * KD);
  edge_scores   <<<(KET + 7) / 8, 256, 0, stream>>>(ei, xl, xr, att, e, emax);
  edge_softmax_z<<<(KET * 2 + 255) / 256, 256, 0, stream>>>(ei, e, emax, z, denom);
  edge_aggregate<<<(KET + 7) / 8, 256, 0, stream>>>(ei, xl, z, denom, gat);
  relu_bias_bf16<<<(KN * KD + 255) / 256, 256, 0, stream>>>(gat, bias_c, hb, (size_t)KN * KD);

  // 4. MLP block 1: LN(h@Wd_a+bd_a) -> relu -> + (h@Wal_a+bal_a)
  dim3 g256(256 / 64, KN / 256);
  wmma_gemm_bias<<<g256, 256, 0, stream>>>(hb, WdaT,  bd_a,  t1, KN, 256, 512);
  wmma_gemm_bias<<<g256, 256, 0, stream>>>(hb, WalaT, bal_a, r1, KN, 256, 512);
  ln_relu_res<<<KN / 8, 256, 0, stream>>>(t1, r1, ga, bta, nullptr, h1b, 256, KN);

  // 5. MLP block 2
  dim3 g128(128 / 64, KN / 256);
  wmma_gemm_bias<<<g128, 256, 0, stream>>>(h1b, Wd1T,  bd1,  t2, KN, 128, 256);
  wmma_gemm_bias<<<g128, 256, 0, stream>>>(h1b, Wal1T, bal1, r2, KN, 128, 256);
  ln_relu_res<<<KN / 8, 256, 0, stream>>>(t2, r2, g1, bt1, nullptr, h2b, 128, KN);

  // 6. MLP block 3 (no residual) then 64->3
  dim3 g64(64 / 64, KN / 256);
  wmma_gemm_bias<<<g64, 256, 0, stream>>>(h2b, Wd2T, bd2, t3, KN, 64, 128);
  ln_relu_res<<<KN / 8, 256, 0, stream>>>(t3, nullptr, g2, bt2, h3f, nullptr, 64, KN);
  gemm64x3<<<(KN + 255) / 256, 256, 0, stream>>>(h3f, Wd3, bd3, y, KN);

  // 7. pairwise distance matrix (HBM-bound 604 MB write)
  dim3 cg(KN / 16, KN / 16), cb(16, 16);
  cdist_kernel<<<cg, cb, 0, stream>>>(y, out, KN);
}
